// MambaTrajEncoder_9096740733535
// MI455X (gfx1250) — compile-verified, matrix operand order checked by reading, not run on hardware
//
#include <hip/hip_runtime.h>
#include <hip/hip_bf16.h>
#include <math.h>

// Problem constants (from reference)
#define NB     4
#define LSEQ   2048
#define R_TOT  8192      // NB * LSEQ token rows
#define TSTEP  128
#define DMOD   256
#define DIN    512       // inner channels
#define DSN    16        // state size
#define DTRK   16        // dt rank
#define DCONV  4
#define NLAY   3
#define XDBL_LD 64       // 48 (DTR+2*DS) padded to 64

typedef __attribute__((ext_vector_type(16))) __bf16 v16bf;
typedef __attribute__((ext_vector_type(8)))  float  v8f;

union FragU { unsigned u[8]; v16bf v; };

__device__ __forceinline__ unsigned short f32_to_bf16_rne(float f) {
  union { float f; unsigned u; } x; x.f = f;
  unsigned r = x.u + 0x7FFFu + ((x.u >> 16) & 1u);
  return (unsigned short)(r >> 16);
}

// Low 32 bits of a flat LDS pointer == workgroup-relative LDS offset
// (ISA 10.2: LDS aperture lives in addr[63:32]; the LDS offset is the low 32 bits).
__device__ __forceinline__ unsigned lds_off_u32(const void* p) {
  return (unsigned)(unsigned long long)p;
}

// gfx1250 async global->LDS copy, tracked by ASYNCcnt (ISA 08 §4).
__device__ __forceinline__ void async_load_b128(unsigned lds_off, const void* gaddr) {
  asm volatile("global_load_async_to_lds_b128 %0, %1, off"
               :: "v"(lds_off), "v"(gaddr) : "memory");
}
__device__ __forceinline__ void wait_async_le3() {
  asm volatile("s_wait_asynccnt 0x3" ::: "memory");
}
__device__ __forceinline__ void wait_async_0() {
  asm volatile("s_wait_asynccnt 0x0" ::: "memory");
}

// ---------------------------------------------------------------------------
// f32 -> bf16 conversion with optional zero row padding (for x_proj 48->64)
// ---------------------------------------------------------------------------
__global__ void convert_pad_bf16(const float* __restrict__ src,
                                 unsigned short* __restrict__ dst,
                                 int rows_src, int rows_dst, int cols) {
  int i = blockIdx.x * blockDim.x + threadIdx.x;
  if (i >= rows_dst * cols) return;
  int r = i / cols;
  int c = i - r * cols;
  float v = (r < rows_src) ? src[(size_t)r * cols + c] : 0.f;
  dst[i] = f32_to_bf16_rne(v);
}

// ---------------------------------------------------------------------------
// bf16 WMMA GEMM: C[M,N] = A[M,K] @ W[N,K]^T (+bias) (+=residual)
// Block tile 128x64; 8 waves in a 4x2 grid, each owning a 32x32 output quad
// (2 A frags x 2 B frags -> 4 WMMAs per k-step). K stepped by 32.
// Double-buffered LDS fed by global_load_async_to_lds_b128 (3 loads/tile/
// thread); s_wait_asynccnt 3 overlaps tile j+1's loads with tile j's WMMAs.
// LDS row pitch 40 halves = 80B: 16B-aligned b128 chunks and conflict-free
// fragment gathers (20-dword stride, 16 rows on 16 distinct banks of 64).
// ---------------------------------------------------------------------------
#define BM 128
#define BN 64
#define BK 32
#define LDT 40

__global__ __launch_bounds__(256) void gemm_bf16_wmma(
    const unsigned short* __restrict__ A,   // [M,K] bf16 row-major
    const unsigned short* __restrict__ W,   // [N,K] bf16 row-major
    float* __restrict__ C,                  // [M,N] f32
    const float* __restrict__ bias,         // [N] or nullptr
    int M, int N, int K, int add_residual)
{
  __shared__ unsigned short sA[2][BM * LDT];
  __shared__ unsigned short sB[2][BN * LDT];
  const int tid  = threadIdx.x;
  const int wave = tid >> 5;
  const int lane = tid & 31;
  const int bm = blockIdx.x * BM;
  const int bn = blockIdx.y * BN;
  const int tm = (wave & 3) * 32;   // wave quad row within block (2 tiles)
  const int tn = (wave >> 2) * 32;  // wave quad col within block (2 tiles)
  const int mrow = lane & 15;
  const int khi  = (lane >> 4) << 3;

  // A staging: 128 rows x 32 cols, two b128 per thread (same row)
  const int axr = tid >> 1;               // 0..127
  const int axc = (tid & 1) * 16;         // 0 or 16
  // B staging: 64 rows x 32 cols, one b128 per thread
  const int bxr = tid >> 2;               // 0..63
  const int bxc = (tid & 3) * 8;          // 0,8,16,24
  const unsigned short* gA = A + (size_t)(bm + axr) * K + axc;
  const unsigned short* gB = W + (size_t)(bn + bxr) * K + bxc;
  const unsigned sAo[2] = { lds_off_u32(&sA[0][axr * LDT + axc]),
                            lds_off_u32(&sA[1][axr * LDT + axc]) };
  const unsigned sBo[2] = { lds_off_u32(&sB[0][bxr * LDT + bxc]),
                            lds_off_u32(&sB[1][bxr * LDT + bxc]) };

  v8f acc00 = {0.f,0.f,0.f,0.f,0.f,0.f,0.f,0.f};
  v8f acc01 = acc00, acc10 = acc00, acc11 = acc00;

  const int ntiles = K / BK;
  // prologue: tile 0 -> buffer 0
  async_load_b128(sAo[0],      gA);
  async_load_b128(sAo[0] + 16, gA + 8);   // +8 halves = +16B in both spaces
  async_load_b128(sBo[0],      gB);

  // fragment K index: VGPR v holds K = 2v + 8*[v>=4] + 8*[lane>=16] .. +1
  // (gathers below are contiguous b128 pairs after this mapping)
#define GEMM_COMPUTE(BUF)                                                     \
  {                                                                           \
    FragU fa0, fa1, fb0, fb1;                                                 \
    _Pragma("unroll")                                                         \
    for (int v = 0; v < 8; ++v) {                                             \
      const int kk = 2 * v + ((v >= 4) ? 8 : 0) + khi;                        \
      fa0.u[v] = *(const unsigned*)&sA[BUF][(tm + mrow) * LDT + kk];          \
      fa1.u[v] = *(const unsigned*)&sA[BUF][(tm + 16 + mrow) * LDT + kk];     \
      fb0.u[v] = *(const unsigned*)&sB[BUF][(tn + mrow) * LDT + kk];          \
      fb1.u[v] = *(const unsigned*)&sB[BUF][(tn + 16 + mrow) * LDT + kk];     \
    }                                                                         \
    acc00 = __builtin_amdgcn_wmma_f32_16x16x32_bf16(false, fa0.v, false,      \
                fb0.v, (short)0, acc00, false, false);                        \
    acc01 = __builtin_amdgcn_wmma_f32_16x16x32_bf16(false, fa0.v, false,      \
                fb1.v, (short)0, acc01, false, false);                        \
    acc10 = __builtin_amdgcn_wmma_f32_16x16x32_bf16(false, fa1.v, false,      \
                fb0.v, (short)0, acc10, false, false);                        \
    acc11 = __builtin_amdgcn_wmma_f32_16x16x32_bf16(false, fa1.v, false,      \
                fb1.v, (short)0, acc11, false, false);                        \
  }

  int j = 0;
  for (; j < ntiles - 1; ++j) {
    const int buf = j & 1;
    // issue tile j+1 into the other buffer (safe: last read of that buffer
    // was fenced by iteration j-1's closing barrier)
    async_load_b128(sAo[buf ^ 1],      gA + (j + 1) * BK);
    async_load_b128(sAo[buf ^ 1] + 16, gA + (j + 1) * BK + 8);
    async_load_b128(sBo[buf ^ 1],      gB + (j + 1) * BK);
    wait_async_le3();   // tile j landed; tile j+1 still in flight
    __syncthreads();
    GEMM_COMPUTE(buf)
    __syncthreads();
  }
  wait_async_0();
  __syncthreads();
  GEMM_COMPUTE(j & 1)

  // D layout: VGPR e -> M = e + 8*[lane>=16], N = lane%16
  const int ncol0 = bn + tn + (lane & 15);
  const int ncol1 = ncol0 + 16;
  const int mb0 = bm + tm + ((lane >> 4) << 3);
  const int mb1 = mb0 + 16;
  const float bv0 = bias ? bias[ncol0] : 0.f;
  const float bv1 = bias ? bias[ncol1] : 0.f;
#pragma unroll
  for (int e = 0; e < 8; ++e) {
    const size_t r0 = (size_t)(mb0 + e);
    const size_t r1 = (size_t)(mb1 + e);
    float v00 = acc00[e] + bv0, v01 = acc01[e] + bv1;
    float v10 = acc10[e] + bv0, v11 = acc11[e] + bv1;
    if (add_residual) {
      v00 += C[r0 * N + ncol0]; v01 += C[r0 * N + ncol1];
      v10 += C[r1 * N + ncol0]; v11 += C[r1 * N + ncol1];
    }
    C[r0 * N + ncol0] = v00; C[r0 * N + ncol1] = v01;
    C[r1 * N + ncol0] = v10; C[r1 * N + ncol1] = v11;
  }
#undef GEMM_COMPUTE
}

// ---------------------------------------------------------------------------
// LayerNorm over DM=256, one wave32 per row (8 f32 per lane), shuffle reduce.
// ---------------------------------------------------------------------------
__global__ __launch_bounds__(256) void layernorm_kernel(
    const float* __restrict__ x, const float* __restrict__ w,
    const float* __restrict__ b,
    unsigned short* __restrict__ out_bf, float* __restrict__ out_f)
{
  const int wave = threadIdx.x >> 5, lane = threadIdx.x & 31;
  const int row = blockIdx.x * 8 + wave;
  const float* xr = x + (size_t)row * DMOD;
  float v[8]; float s = 0.f;
#pragma unroll
  for (int i = 0; i < 8; ++i) { v[i] = xr[lane + 32 * i]; s += v[i]; }
#pragma unroll
  for (int m = 16; m >= 1; m >>= 1) s += __shfl_xor(s, m, 32);
  const float mu = s * (1.f / 256.f);
  float q = 0.f;
#pragma unroll
  for (int i = 0; i < 8; ++i) { float d = v[i] - mu; q += d * d; }
#pragma unroll
  for (int m = 16; m >= 1; m >>= 1) q += __shfl_xor(q, m, 32);
  const float rstd = rsqrtf(q * (1.f / 256.f) + 1e-5f);
#pragma unroll
  for (int i = 0; i < 8; ++i) {
    const int c = lane + 32 * i;
    const float o = (v[i] - mu) * rstd * w[c] + b[c];
    const size_t idx = (size_t)row * DMOD + c;
    if (out_bf) out_bf[idx] = f32_to_bf16_rne(o);
    if (out_f)  out_f[idx]  = o;
  }
}

// ---------------------------------------------------------------------------
// Causal depthwise conv (DC=4) + SiLU over xm half of xz; emits f32 + bf16.
// ---------------------------------------------------------------------------
__global__ void conv_silu_kernel(const float* __restrict__ xz,
                                 const float* __restrict__ cw,
                                 const float* __restrict__ cb,
                                 float* __restrict__ xc,
                                 unsigned short* __restrict__ xc_bf)
{
  const int i = blockIdx.x * blockDim.x + threadIdx.x;  // over R_TOT*DIN
  const int d = i & (DIN - 1);
  const int r = i >> 9;
  const int t = r & (LSEQ - 1);
  float acc = cb[d];
  const float* w = cw + d * DCONV;
#pragma unroll
  for (int j = 0; j < DCONV; ++j) {
    const int tt = t - (DCONV - 1) + j;
    if (tt >= 0)
      acc += w[j] * xz[(size_t)(r - (DCONV - 1) + j) * (2 * DIN) + d];
  }
  const float sv = acc * (1.f / (1.f + __expf(-acc)));
  xc[i] = sv;
  xc_bf[i] = f32_to_bf16_rne(sv);
}

// ---------------------------------------------------------------------------
// delta = softplus(dt @ dt_proj_w^T + b); K=16 -> cheap VALU dot.
// ---------------------------------------------------------------------------
__global__ void delta_kernel(const float* __restrict__ x_dbl,
                             const float* __restrict__ dtw,
                             const float* __restrict__ dtb,
                             float* __restrict__ delta)
{
  const int i = blockIdx.x * blockDim.x + threadIdx.x;
  const int d = i & (DIN - 1);
  const int r = i >> 9;
  const float* xr = x_dbl + (size_t)r * XDBL_LD;
  const float* w  = dtw + d * DTRK;
  float acc = dtb[d];
#pragma unroll
  for (int j = 0; j < DTRK; ++j) acc += xr[j] * w[j];
  delta[i] = (acc > 20.f) ? acc : log1pf(__expf(acc));
}

// ---------------------------------------------------------------------------
// Selective scan with fused exp(delta*A), delta*B*u (never materializes the
// [B,L,DI,DS] tensors -> saves ~0.5GB traffic, keeps data L2-resident).
// One wave = 2 channels x 16 states; Σ_n h·C via 4 shfl_xor hops.
// ---------------------------------------------------------------------------
__global__ __launch_bounds__(256) void scan_kernel(
    const float* __restrict__ delta, const float* __restrict__ x_dbl,
    const float* __restrict__ xc, const float* __restrict__ A_log,
    float* __restrict__ y)
{
  const int wave = threadIdx.x >> 5, lane = threadIdx.x & 31;
  const int b = blockIdx.x >> 5;
  const int dblk = (blockIdx.x & 31) * 16;
  const int n = lane & 15;
  const int d = dblk + wave * 2 + (lane >> 4);
  const float A = -__expf(A_log[d * DSN + n]);
  float h = 0.f;
  const size_t rbase = (size_t)b * LSEQ;
  for (int t = 0; t < LSEQ; ++t) {
    const size_t r = rbase + t;
    if (t + 1 < LSEQ) {
      __builtin_prefetch(&delta[(r + 1) * DIN + d], 0, 1);
      __builtin_prefetch(&x_dbl[(r + 1) * XDBL_LD + DTRK + n], 0, 1);
      __builtin_prefetch(&xc[(r + 1) * DIN + d], 0, 1);
    }
    const float dl = delta[r * DIN + d];
    const float Bm = x_dbl[r * XDBL_LD + DTRK + n];
    const float Cm = x_dbl[r * XDBL_LD + DTRK + DSN + n];
    const float u  = xc[r * DIN + d];
    h = __expf(dl * A) * h + (dl * Bm) * u;
    float v = h * Cm;
    v += __shfl_xor(v, 1, 32);
    v += __shfl_xor(v, 2, 32);
    v += __shfl_xor(v, 4, 32);
    v += __shfl_xor(v, 8, 32);
    if (n == 0) y[r * DIN + d] = v;
  }
}

// ---------------------------------------------------------------------------
// y = (scan + xc*D) * silu(z) -> bf16 for out_proj GEMM
// ---------------------------------------------------------------------------
__global__ void gate_kernel(const float* __restrict__ y,
                            const float* __restrict__ xc,
                            const float* __restrict__ xz,
                            const float* __restrict__ Dskip,
                            unsigned short* __restrict__ yg_bf)
{
  const int i = blockIdx.x * blockDim.x + threadIdx.x;
  const int d = i & (DIN - 1);
  const int r = i >> 9;
  const float z = xz[(size_t)r * (2 * DIN) + DIN + d];
  const float g = z * (1.f / (1.f + __expf(-z)));
  yg_bf[i] = f32_to_bf16_rne((y[i] + xc[i] * Dskip[d]) * g);
}

// ---------------------------------------------------------------------------
extern "C" void kernel_launch(void* const* d_in, const int* in_sizes, int n_in,
                              void* d_out, int out_size, void* d_ws, size_t ws_size,
                              hipStream_t stream) {
  (void)in_sizes; (void)n_in; (void)out_size; (void)ws_size;
  const float* seq        = (const float*)d_in[0];
  const float* inp_w      = (const float*)d_in[1];
  const float* inp_b      = (const float*)d_in[2];
  const float* ln_w       = (const float*)d_in[3];
  const float* ln_b       = (const float*)d_in[4];
  const float* in_proj_w  = (const float*)d_in[5];
  const float* conv_w     = (const float*)d_in[6];
  const float* conv_b     = (const float*)d_in[7];
  const float* x_proj_w   = (const float*)d_in[8];
  const float* dt_proj_w  = (const float*)d_in[9];
  const float* dt_proj_b  = (const float*)d_in[10];
  const float* A_log      = (const float*)d_in[11];
  const float* D_skip     = (const float*)d_in[12];
  const float* out_proj_w = (const float*)d_in[13];
  const float* out_ln_w   = (const float*)d_in[14];
  const float* out_ln_b   = (const float*)d_in[15];
  float* out = (float*)d_out;

  char* ws = (char*)d_ws;
  size_t off = 0;
  auto alloc = [&](size_t bytes) -> void* {
    void* p = ws + off;
    off += (bytes + 255) & ~(size_t)255;
    return p;
  };

  float*          h       = (float*)alloc((size_t)R_TOT * DMOD * 4);
  unsigned short* seq_bf  = (unsigned short*)alloc((size_t)R_TOT * TSTEP * 2);
  unsigned short* ln_bf   = (unsigned short*)alloc((size_t)R_TOT * DMOD * 2);
  float*          xz      = (float*)alloc((size_t)R_TOT * 2 * DIN * 4);
  float*          xc      = (float*)alloc((size_t)R_TOT * DIN * 4);
  unsigned short* xc_bf   = (unsigned short*)alloc((size_t)R_TOT * DIN * 2);
  float*          x_dbl   = (float*)alloc((size_t)R_TOT * XDBL_LD * 4);
  float*          dlt     = (float*)alloc((size_t)R_TOT * DIN * 4);
  float*          ysc     = (float*)alloc((size_t)R_TOT * DIN * 4);
  unsigned short* yg_bf   = (unsigned short*)alloc((size_t)R_TOT * DIN * 2);
  unsigned short* wbf_inp     = (unsigned short*)alloc((size_t)DMOD * TSTEP * 2);
  unsigned short* wbf_inproj  = (unsigned short*)alloc((size_t)NLAY * 2 * DIN * DMOD * 2);
  unsigned short* wbf_xproj   = (unsigned short*)alloc((size_t)NLAY * XDBL_LD * DIN * 2);
  unsigned short* wbf_outproj = (unsigned short*)alloc((size_t)NLAY * DMOD * DIN * 2);

  const int CT = 256;
  auto cdiv = [](int a, int b) { return (a + b - 1) / b; };

  convert_pad_bf16<<<cdiv(R_TOT * TSTEP, CT), CT, 0, stream>>>(seq, seq_bf, R_TOT, R_TOT, TSTEP);
  convert_pad_bf16<<<cdiv(DMOD * TSTEP, CT), CT, 0, stream>>>(inp_w, wbf_inp, DMOD, DMOD, TSTEP);
  for (int l = 0; l < NLAY; ++l) {
    convert_pad_bf16<<<cdiv(2 * DIN * DMOD, CT), CT, 0, stream>>>(
        in_proj_w + (size_t)l * 2 * DIN * DMOD, wbf_inproj + (size_t)l * 2 * DIN * DMOD,
        2 * DIN, 2 * DIN, DMOD);
    convert_pad_bf16<<<cdiv(XDBL_LD * DIN, CT), CT, 0, stream>>>(
        x_proj_w + (size_t)l * 48 * DIN, wbf_xproj + (size_t)l * XDBL_LD * DIN,
        48, XDBL_LD, DIN);
    convert_pad_bf16<<<cdiv(DMOD * DIN, CT), CT, 0, stream>>>(
        out_proj_w + (size_t)l * DMOD * DIN, wbf_outproj + (size_t)l * DMOD * DIN,
        DMOD, DMOD, DIN);
  }

  // h = seq @ inp_w^T + inp_b
  gemm_bf16_wmma<<<dim3(R_TOT / BM, DMOD / BN), 256, 0, stream>>>(
      seq_bf, wbf_inp, h, inp_b, R_TOT, DMOD, TSTEP, 0);

  for (int l = 0; l < NLAY; ++l) {
    layernorm_kernel<<<R_TOT / 8, 256, 0, stream>>>(
        h, ln_w + l * DMOD, ln_b + l * DMOD, ln_bf, nullptr);
    gemm_bf16_wmma<<<dim3(R_TOT / BM, (2 * DIN) / BN), 256, 0, stream>>>(
        ln_bf, wbf_inproj + (size_t)l * 2 * DIN * DMOD, xz, nullptr,
        R_TOT, 2 * DIN, DMOD, 0);
    conv_silu_kernel<<<cdiv(R_TOT * DIN, CT), CT, 0, stream>>>(
        xz, conv_w + l * DIN * DCONV, conv_b + l * DIN, xc, xc_bf);
    gemm_bf16_wmma<<<dim3(R_TOT / BM, XDBL_LD / BN), 256, 0, stream>>>(
        xc_bf, wbf_xproj + (size_t)l * XDBL_LD * DIN, x_dbl, nullptr,
        R_TOT, XDBL_LD, DIN, 0);
    delta_kernel<<<cdiv(R_TOT * DIN, CT), CT, 0, stream>>>(
        x_dbl, dt_proj_w + l * DIN * DTRK, dt_proj_b + l * DIN, dlt);
    scan_kernel<<<NB * 32, 256, 0, stream>>>(
        dlt, x_dbl, xc, A_log + l * DIN * DSN, ysc);
    gate_kernel<<<cdiv(R_TOT * DIN, CT), CT, 0, stream>>>(
        ysc, xc, xz, D_skip + l * DIN, yg_bf);
    gemm_bf16_wmma<<<dim3(R_TOT / BM, DMOD / BN), 256, 0, stream>>>(
        yg_bf, wbf_outproj + (size_t)l * DMOD * DIN, h, nullptr,
        R_TOT, DMOD, DIN, 1);
  }

  layernorm_kernel<<<R_TOT / 8, 256, 0, stream>>>(
      h, out_ln_w, out_ln_b, nullptr, out);
}
